// LiftingModule_89446988907042
// MI455X (gfx1250) — compile-verified, matrix operand order checked by reading, not run on hardware
//
#include <hip/hip_runtime.h>
#include <hip/hip_bf16.h>
#include <math.h>

// ---------------- types ----------------
typedef __bf16 bf16_t;
typedef __attribute__((ext_vector_type(16))) __bf16 bf16x16;
typedef __attribute__((ext_vector_type(8)))  float  f32x8;
typedef __attribute__((ext_vector_type(4)))  unsigned int u32x4;
typedef int v4i_vec __attribute__((vector_size(16)));   // matches builtin param pointee

#define AS_GLOBAL __attribute__((address_space(1)))
#define AS_LDS    __attribute__((address_space(3)))

// async global->LDS staging path (CDNA5), guarded so compile never breaks
#if defined(__gfx1250__) && __has_builtin(__builtin_amdgcn_global_load_async_to_lds_b128)
#define ASYNC_STAGE 1
#endif

// ---------------- problem constants ----------------
constexpr int D_   = 256;
constexpr int KPTS = 4;
constexpr int SX_ = 24, SY_ = 24, SZ_ = 24;
constexpr int NQ  = SX_ * SY_ * SZ_;      // 13824 queries
constexpr int CC  = 6;                    // cameras
constexpr int HF_ = 28, WF_ = 50;
constexpr int IMGH = 896, IMGW = 1600;

// ================= init: queries = query_base + pos_enc =================
__global__ __launch_bounds__(256) void init_queries_kernel(
    const float* __restrict__ qbase, float* __restrict__ queries)
{
    int idx = blockIdx.x * blockDim.x + threadIdx.x;
    if (idx >= NQ * D_) return;
    int n = idx >> 8, d = idx & 255;
    int ix = n / (SY_ * SZ_), iy = (n / SZ_) % SY_, iz = n % SZ_;
    float coord[3];
    coord[0] = -1.f + 2.f * ix / (SX_ - 1);
    coord[1] = -1.f + 2.f * iy / (SY_ - 1);
    coord[2] = -1.f + 2.f * iz / (SZ_ - 1);
    float val = 0.f;
    if (d < 252) {                      // nf = 42, 3 axes * 84
        int a = d / 84, m = d % 84;
        int j = (m < 42) ? m : m - 42;
        float freq = __powf(10000.f, -(float)j / 42.f);
        float ang = coord[a] * freq;
        val = (m < 42) ? __sinf(ang) : __cosf(ang);
    }
    queries[idx] = qbase[d] + val;
}

// ================= camera projection =================
__global__ __launch_bounds__(256) void project_kernel(
    const float* __restrict__ intr, const float* __restrict__ extr,
    float* __restrict__ pc, float* __restrict__ vm)
{
    int idx = blockIdx.x * blockDim.x + threadIdx.x;
    if (idx >= CC * NQ) return;
    int c = idx / NQ, n = idx % NQ;
    int ix = n / (SY_ * SZ_), iy = (n / SZ_) % SY_, iz = n % SZ_;
    float px = -1.f + 2.f * ix / (SX_ - 1);
    float py = -1.f + 2.f * iy / (SY_ - 1);
    float pz = -1.f + 2.f * iz / (SZ_ - 1);
    const float* E = extr + c * 16;
    float cam[3];
#pragma unroll
    for (int i = 0; i < 3; ++i)
        cam[i] = E[i*4+0]*px + E[i*4+1]*py + E[i*4+2]*pz + E[i*4+3];
    float z = cam[2];
    const float* Ki = intr + c * 9;
    float u = Ki[0]*cam[0] + Ki[1]*cam[1] + Ki[2]*cam[2];
    float v = Ki[3]*cam[0] + Ki[4]*cam[1] + Ki[5]*cam[2];
    float zs = fmaxf(z, 1e-3f);
    float xn = 2.f * (u / zs) / (IMGW - 1) - 1.f;
    float yn = 2.f * (v / zs) / (IMGH - 1) - 1.f;
    bool valid = (z > 1e-3f) && (fabsf(xn) <= 1.f) && (fabsf(yn) <= 1.f);
    pc[idx*2+0] = xn; pc[idx*2+1] = yn;
    vm[idx] = valid ? 1.f : 0.f;
}

// ================= LayerNorm (one wave32 per row of 256) =================
__global__ __launch_bounds__(256) void ln_kernel(
    const float* __restrict__ x, const float* __restrict__ g,
    const float* __restrict__ b, bf16_t* __restrict__ y, int nrows)
{
    int warp = threadIdx.x >> 5, lane = threadIdx.x & 31;
    int row = blockIdx.x * 8 + warp;
    if (row >= nrows) return;
    const float* xr = x + (size_t)row * D_ + lane * 8;
    float v[8], s = 0.f, ss = 0.f;
#pragma unroll
    for (int j = 0; j < 8; ++j) { v[j] = xr[j]; s += v[j]; ss += v[j]*v[j]; }
#pragma unroll
    for (int m = 16; m >= 1; m >>= 1) { s += __shfl_xor(s, m, 32); ss += __shfl_xor(ss, m, 32); }
    float mean = s * (1.f / D_);
    float var  = ss * (1.f / D_) - mean * mean;
    float rstd = rsqrtf(var + 1e-5f);
    bf16_t* yr = y + (size_t)row * D_ + lane * 8;
#pragma unroll
    for (int j = 0; j < 8; ++j) {
        int d = lane * 8 + j;
        yr[j] = (bf16_t)((v[j] - mean) * rstd * g[d] + b[d]);
    }
}

// ================= f32 -> bf16 convert =================
__global__ __launch_bounds__(256) void f32_to_bf16_kernel(
    const float* __restrict__ src, bf16_t* __restrict__ dst, int n)
{
    int i = blockIdx.x * blockDim.x + threadIdx.x;
    if (i < n) dst[i] = (bf16_t)src[i];
}

// ================= zero =================
__global__ __launch_bounds__(256) void zero_kernel(float* __restrict__ p, int n)
{
    int i = blockIdx.x * blockDim.x + threadIdx.x;
    if (i < n) p[i] = 0.f;
}

// ================= small linear for K*2 offsets (Nout=8) =================
__global__ __launch_bounds__(256) void offset_linear_kernel(
    const bf16_t* __restrict__ qn, const float* __restrict__ w,
    const float* __restrict__ b, float* __restrict__ off, int nrows)
{
    int n = blockIdx.x * blockDim.x + threadIdx.x;
    if (n >= nrows) return;
    const bf16_t* xr = qn + (size_t)n * D_;
    float acc[8];
#pragma unroll
    for (int o = 0; o < 8; ++o) acc[o] = b[o];
    for (int d = 0; d < D_; ++d) {
        float x = (float)xr[d];
#pragma unroll
        for (int o = 0; o < 8; ++o) acc[o] += x * w[o * D_ + d];
    }
#pragma unroll
    for (int o = 0; o < 8; ++o) off[n * 8 + o] = acc[o] * 0.05f;
}

// ================= bilinear sampling (block = one (n,c); thread = channel) =====
// off != nullptr: 4 offset points, mean (per-image layer). off == nullptr: 1 point.
__global__ __launch_bounds__(256) void sample_kernel(
    const float* __restrict__ feat, const float* __restrict__ pc,
    const float* __restrict__ off, bf16_t* __restrict__ sm)
{
    int n = blockIdx.x, c = blockIdx.y, d = threadIdx.x;
    size_t cn = (size_t)c * NQ + n;
    float bx = pc[cn * 2 + 0], by = pc[cn * 2 + 1];
    int nk = off ? KPTS : 1;
    float wk = off ? (1.f / KPTS) : 1.f;
    float acc = 0.f;
    for (int k = 0; k < nk; ++k) {
        float cx = bx, cy = by;
        if (off) { cx += off[(n * KPTS + k) * 2 + 0]; cy += off[(n * KPTS + k) * 2 + 1]; }
        float x = (cx + 1.f) * 0.5f * (WF_ - 1);
        float y = (cy + 1.f) * 0.5f * (HF_ - 1);
        float x0 = floorf(x), y0 = floorf(y);
        float wx1 = x - x0, wy1 = y - y0;
#pragma unroll
        for (int dy = 0; dy < 2; ++dy) {
#pragma unroll
            for (int dx = 0; dx < 2; ++dx) {
                float xf = x0 + dx, yf = y0 + dy;
                float w = (dx ? wx1 : 1.f - wx1) * (dy ? wy1 : 1.f - wy1);
                if (xf >= 0.f && xf < (float)WF_ && yf >= 0.f && yf < (float)HF_) {
                    int xi = (int)xf, yi = (int)yf;
                    acc += w * feat[(((size_t)c * HF_ + yi) * WF_ + xi) * D_ + d];
                }
            }
        }
    }
    sm[cn * D_ + d] = (bf16_t)(acc * wk);
}

// ================= WMMA bf16 GEMM: C[M,256] (+)= A[M,256] @ W[256,256]^T + bias =====
// A row-major (M x 256) bf16, W row-major (256 x 256) bf16 (rows = output features).
// Block: 256 threads = 8 waves; wave -> 16x64 tile; grid = (M/128, 4).
// The 64 W rows used by a block are staged once into LDS (async copy on CDNA5),
// and all A fragments are preloaded so the hot loop is ds_load + v_wmma only.
__device__ __forceinline__ bf16x16 load_frag(const bf16_t* p)
{
    union { bf16x16 v; u32x4 q[2]; } u;
    u.q[0] = *(const u32x4*)(p);        // K elements e=0..7
    u.q[1] = *(const u32x4*)(p + 16);   // K elements e=8..15
    return u.v;
}

constexpr int WPAD    = 8;          // 16B pad per row: spreads LDS banks
constexpr int WSTRIDE = 256 + WPAD; // elements per staged W row

__global__ __launch_bounds__(256) void gemm256_kernel(
    const bf16_t* __restrict__ A, const bf16_t* __restrict__ W,
    const float* __restrict__ bias, float* __restrict__ C,
    int M, int accumulate)
{
    __shared__ bf16_t wlds[64 * WSTRIDE];

    int lane = threadIdx.x & 31, wave = threadIdx.x >> 5;
    int row0 = (blockIdx.x * 8 + wave) * 16;      // M always multiple of 128
    int col0 = blockIdx.y * 64;

    // ---- stage 64 weight rows (this block's output cols) into LDS ----
    {
        int t = threadIdx.x;
#pragma unroll
        for (int i = 0; i < 8; ++i) {
            int chunk = i * 256 + t;              // 2048 x 16B chunks, coalesced
            int r  = chunk >> 5;                  // 32 chunks per row
            int ko = (chunk & 31) * 8;            // element offset in row
            const bf16_t* src = W + (size_t)(col0 + r) * 256 + ko;
            bf16_t* dst = &wlds[r * WSTRIDE + ko];
#if defined(ASYNC_STAGE)
            __builtin_amdgcn_global_load_async_to_lds_b128(
                (AS_GLOBAL v4i_vec*)(void*)const_cast<bf16_t*>(src),
                (AS_LDS v4i_vec*)dst, 0, 0);
#else
            *(u32x4*)dst = *(const u32x4*)src;
#endif
        }
#if defined(ASYNC_STAGE)
#if __has_builtin(__builtin_amdgcn_s_wait_asynccnt)
        __builtin_amdgcn_s_wait_asynccnt(0);
#else
        asm volatile("s_wait_asynccnt 0" ::: "memory");
#endif
#endif
    }

    // ---- preload all 8 A k-fragments while staging completes ----
    int half = lane >> 4, r = lane & 15;
    const bf16_t* abase = A + (size_t)(row0 + r) * 256 + half * 8;
    bf16x16 af[8];
#pragma unroll
    for (int i = 0; i < 8; ++i) af[i] = load_frag(abase + i * 32);

    __syncthreads();

    // ---- hot loop: LDS fragment reads + WMMA ----
    f32x8 acc[4] = {};
#pragma unroll
    for (int i = 0; i < 8; ++i) {
        int kb = i * 32;
#pragma unroll
        for (int j = 0; j < 4; ++j) {
            const bf16_t* wp = &wlds[(j * 16 + r) * WSTRIDE + kb + half * 8];
            bf16x16 bfg = load_frag(wp);
            acc[j] = __builtin_amdgcn_wmma_f32_16x16x32_bf16(
                false, af[i], false, bfg, (short)0, acc[j], false, false);
        }
    }

#pragma unroll
    for (int j = 0; j < 4; ++j) {
        int col = col0 + j * 16 + r;
        float bv = bias ? bias[col] : 0.f;
#pragma unroll
        for (int e = 0; e < 8; ++e) {
            int m = row0 + half * 8 + e;
            float val = acc[j][e] + bv;
            float* cp = C + (size_t)m * 256 + col;
            if (accumulate) *cp += val; else *cp = val;
        }
    }
}

// ================= row dot product: s = (q . k)/16 [* mask] =================
__global__ __launch_bounds__(256) void rowdot_kernel(
    const float* __restrict__ q, const float* __restrict__ k,
    const float* __restrict__ mask, float* __restrict__ out,
    int out_stride, int out_off, int nrows)
{
    int warp = threadIdx.x >> 5, lane = threadIdx.x & 31;
    int row = blockIdx.x * 8 + warp;
    if (row >= nrows) return;
    const float* qr = q + (size_t)row * D_ + lane * 8;
    const float* kr = k + (size_t)row * D_ + lane * 8;
    float s = 0.f;
#pragma unroll
    for (int j = 0; j < 8; ++j) s += qr[j] * kr[j];
#pragma unroll
    for (int m = 16; m >= 1; m >>= 1) s += __shfl_xor(s, m, 32);
    if (lane == 0) {
        float v = s * (1.f / 16.f);       // / sqrt(256)
        if (mask) v *= mask[row];
        out[(size_t)row * out_stride + out_off] = v;
    }
}

// ================= agg[n,d] += s[n*stride+off] * v[n,d] =================
__global__ __launch_bounds__(256) void axpy_rows_kernel(
    float* __restrict__ agg, const float* __restrict__ s, int s_stride, int s_off,
    const float* __restrict__ v, int total)
{
    int i = blockIdx.x * blockDim.x + threadIdx.x;
    if (i >= total) return;
    int n = i >> 8;
    agg[i] += s[(size_t)n * s_stride + s_off] * v[i];
}

// ================= per-image finalize: agg/tw -> bf16 =================
__global__ __launch_bounds__(256) void finalize_agg_kernel(
    const float* __restrict__ agg, const float* __restrict__ vm,
    bf16_t* __restrict__ out, int total)
{
    int i = blockIdx.x * blockDim.x + threadIdx.x;
    if (i >= total) return;
    int n = i >> 8;
    float tw = 0.f;
#pragma unroll
    for (int c = 0; c < CC; ++c) tw += vm[(size_t)c * NQ + n];
    tw = fmaxf(tw, 1.f);
    out[i] = (bf16_t)(agg[i] / tw);
}

// ================= masked softmax over cameras (in place) =================
__global__ __launch_bounds__(256) void softmax_kernel(
    float* __restrict__ logits, const float* __restrict__ vm, int nrows)
{
    int n = blockIdx.x * blockDim.x + threadIdx.x;
    if (n >= nrows) return;
    float l[CC]; bool vd[CC]; bool any = false; float m = -1e30f;
#pragma unroll
    for (int c = 0; c < CC; ++c) {
        l[c] = logits[(size_t)n * CC + c];
        vd[c] = vm[(size_t)c * NQ + n] > 0.5f;
        if (vd[c]) { any = true; m = fmaxf(m, l[c]); }
    }
    float e[CC]; float sum = 0.f;
#pragma unroll
    for (int c = 0; c < CC; ++c) {
        e[c] = (any && vd[c]) ? expf(l[c] - m) : 0.f;
        sum += e[c];
    }
    float inv = any ? (1.f / sum) : 0.f;
#pragma unroll
    for (int c = 0; c < CC; ++c) logits[(size_t)n * CC + c] = e[c] * inv;
}

// ================= final axis means =================
__global__ __launch_bounds__(256) void reduce_kernel(
    const float* __restrict__ q, float* __restrict__ out)
{
    constexpr int SEC = SX_ * SY_ * D_;   // 147456
    int idx = blockIdx.x * blockDim.x + threadIdx.x;
    if (idx >= 3 * SEC) return;
    int sec = idx / SEC, r = idx % SEC;
    int i0 = r / (24 * 256), i1 = (r / 256) % 24, d = r & 255;
    int base, stride;
    if (sec == 0)      { base = (i0 * 24 + i1) * 24; stride = 1;   }  // mean over z
    else if (sec == 1) { base = i0 * 576 + i1;       stride = 24;  }  // mean over y
    else               { base = i0 * 24 + i1;        stride = 576; }  // mean over x
    float s = 0.f;
#pragma unroll
    for (int t = 0; t < 24; ++t) s += q[((size_t)(base + t * stride)) * D_ + d];
    out[idx] = s * (1.f / 24.f);
}

// ================================ host ================================
extern "C" void kernel_launch(void* const* d_in, const int* in_sizes, int n_in,
                              void* d_out, int out_size, void* d_ws, size_t ws_size,
                              hipStream_t stream)
{
    (void)in_sizes; (void)n_in; (void)out_size; (void)ws_size;

    // ---- input indices (dict insertion-order flatten) ----
    const float* feat  = (const float*)d_in[0];   // (1,6,28,50,256)
    const float* intr  = (const float*)d_in[1];   // (1,6,3,3)
    const float* extr  = (const float*)d_in[2];   // (1,6,4,4)
    const float* qbase = (const float*)d_in[3];   // (1,1,256)
    auto pi = [&](int layer, int k) -> const float* { return (const float*)d_in[4  + layer * 12 + k]; };
    auto cr = [&](int layer, int k) -> const float* { return (const float*)d_in[28 + layer * 8  + k]; };
    // per-image keys: 0 q_w 1 q_b 2 k_w 3 k_b 4 v_w 5 v_b 6 out_w 7 out_b 8 off_w 9 off_b 10 ln_g 11 ln_b
    // cross keys:     0 q_w 1 q_b 2 kv_w 3 kv_b 4 out_w 5 out_b 6 ln_g 7 ln_b

    // ---- workspace layout (256B aligned) ----
    size_t cursor = 0;
    auto walloc = [&](size_t bytes) -> char* {
        cursor = (cursor + 255) & ~(size_t)255;
        char* p = (char*)d_ws + cursor;
        cursor += bytes;
        return p;
    };
    float*  queries = (float*) walloc((size_t)NQ * D_ * 4);        // persistent state
    bf16_t* qn_bf   = (bf16_t*)walloc((size_t)NQ * D_ * 2);
    float*  pc      = (float*) walloc((size_t)CC * NQ * 2 * 4);
    float*  vm      = (float*) walloc((size_t)CC * NQ * 4);
    float*  offbuf  = (float*) walloc((size_t)NQ * KPTS * 2 * 4);
    bf16_t* sm_bf   = (bf16_t*)walloc((size_t)CC * NQ * D_ * 2);
    float*  qbuf    = (float*) walloc((size_t)NQ * D_ * 4);
    float*  kvtmp   = (float*) walloc((size_t)NQ * D_ * 4);
    float*  scal    = (float*) walloc((size_t)NQ * CC * 4);
    float*  agg     = (float*) walloc((size_t)NQ * D_ * 4);
    bf16_t* agg_bf  = (bf16_t*)walloc((size_t)NQ * D_ * 2);
    bf16_t* wpool   = (bf16_t*)walloc((size_t)1048576 * 2);

    const int ND = NQ * D_;
    dim3 gemmGrid(NQ / 128, 4);

    auto cvt = [&](const float* src, bf16_t* dst, int n) {
        f32_to_bf16_kernel<<<(n + 255) / 256, 256, 0, stream>>>(src, dst, n);
    };

    // ---- convert weights to bf16 ----
    bf16_t* w_pi_q[2]; bf16_t* w_pi_k[2]; bf16_t* w_pi_v[2]; bf16_t* w_pi_o[2];
    bf16_t* w_cr_q[2]; bf16_t* w_cr_kv[2]; bf16_t* w_cr_o[2];
    for (int i = 0; i < 2; ++i) {
        size_t b = (size_t)i * 262144;
        w_pi_q[i] = wpool + b; w_pi_k[i] = wpool + b + 65536;
        w_pi_v[i] = wpool + b + 131072; w_pi_o[i] = wpool + b + 196608;
        cvt(pi(i, 0), w_pi_q[i], 65536);
        cvt(pi(i, 2), w_pi_k[i], 65536);
        cvt(pi(i, 4), w_pi_v[i], 65536);
        cvt(pi(i, 6), w_pi_o[i], 65536);
        size_t cb = 524288 + (size_t)i * 262144;
        w_cr_q[i] = wpool + cb; w_cr_kv[i] = wpool + cb + 65536;
        w_cr_o[i] = wpool + cb + 196608;
        cvt(cr(i, 0), w_cr_q[i], 65536);
        cvt(cr(i, 2), w_cr_kv[i], 131072);
        cvt(cr(i, 4), w_cr_o[i], 65536);
    }

    // ---- queries + projection ----
    init_queries_kernel<<<(ND + 255) / 256, 256, 0, stream>>>(qbase, queries);
    project_kernel<<<(CC * NQ + 255) / 256, 256, 0, stream>>>(intr, extr, pc, vm);

    // ---- per-image layers ----
    for (int li = 0; li < 2; ++li) {
        ln_kernel<<<NQ / 8, 256, 0, stream>>>(queries, pi(li, 10), pi(li, 11), qn_bf, NQ);
        offset_linear_kernel<<<(NQ + 255) / 256, 256, 0, stream>>>(qn_bf, pi(li, 8), pi(li, 9), offbuf, NQ);
        sample_kernel<<<dim3(NQ, CC), 256, 0, stream>>>(feat, pc, offbuf, sm_bf);
        gemm256_kernel<<<gemmGrid, 256, 0, stream>>>(qn_bf, w_pi_q[li], pi(li, 1), qbuf, NQ, 0);
        zero_kernel<<<(ND + 255) / 256, 256, 0, stream>>>(agg, ND);
        for (int c = 0; c < CC; ++c) {
            const bf16_t* smc = sm_bf + (size_t)c * NQ * D_;
            gemm256_kernel<<<gemmGrid, 256, 0, stream>>>(smc, w_pi_k[li], pi(li, 3), kvtmp, NQ, 0);
            rowdot_kernel<<<NQ / 8, 256, 0, stream>>>(qbuf, kvtmp, vm + (size_t)c * NQ, scal, 1, 0, NQ);
            gemm256_kernel<<<gemmGrid, 256, 0, stream>>>(smc, w_pi_v[li], pi(li, 5), kvtmp, NQ, 0);
            axpy_rows_kernel<<<(ND + 255) / 256, 256, 0, stream>>>(agg, scal, 1, 0, kvtmp, ND);
        }
        finalize_agg_kernel<<<(ND + 255) / 256, 256, 0, stream>>>(agg, vm, agg_bf, ND);
        gemm256_kernel<<<gemmGrid, 256, 0, stream>>>(agg_bf, w_pi_o[li], pi(li, 7), queries, NQ, 1);
    }

    // ---- cross-image layers ----
    for (int li = 0; li < 2; ++li) {
        ln_kernel<<<NQ / 8, 256, 0, stream>>>(queries, cr(li, 6), cr(li, 7), qn_bf, NQ);
        sample_kernel<<<dim3(NQ, CC), 256, 0, stream>>>(feat, pc, nullptr, sm_bf);
        gemm256_kernel<<<gemmGrid, 256, 0, stream>>>(qn_bf, w_cr_q[li], cr(li, 1), qbuf, NQ, 0);
        for (int c = 0; c < CC; ++c) {
            const bf16_t* smc = sm_bf + (size_t)c * NQ * D_;
            gemm256_kernel<<<gemmGrid, 256, 0, stream>>>(smc, w_cr_kv[li], cr(li, 3), kvtmp, NQ, 0);
            rowdot_kernel<<<NQ / 8, 256, 0, stream>>>(qbuf, kvtmp, nullptr, scal, CC, c, NQ);
        }
        softmax_kernel<<<(NQ + 255) / 256, 256, 0, stream>>>(scal, vm, NQ);
        zero_kernel<<<(ND + 255) / 256, 256, 0, stream>>>(agg, ND);
        for (int c = 0; c < CC; ++c) {
            const bf16_t* smc = sm_bf + (size_t)c * NQ * D_;
            gemm256_kernel<<<gemmGrid, 256, 0, stream>>>(smc, w_cr_kv[li] + 65536, cr(li, 3) + D_, kvtmp, NQ, 0);
            axpy_rows_kernel<<<(ND + 255) / 256, 256, 0, stream>>>(agg, scal, CC, c, kvtmp, ND);
        }
        f32_to_bf16_kernel<<<(ND + 255) / 256, 256, 0, stream>>>(agg, agg_bf, ND);
        gemm256_kernel<<<gemmGrid, 256, 0, stream>>>(agg_bf, w_cr_o[li], cr(li, 5), queries, NQ, 1);
    }

    // ---- final axis means -> d_out (3 x 24 x 24 x 256 f32, concatenated) ----
    reduce_kernel<<<(3 * SX_ * SY_ * D_ + 255) / 256, 256, 0, stream>>>(queries, (float*)d_out);
}
